// SelfAttention_14431090114896
// MI455X (gfx1250) — matrix operand order repaired and checked by reading, not compiled, not run on hardware
//
#include <hip/hip_runtime.h>

#define EMB 1024
#define SEQ 1024
#define NBATCH 16
#define NHEAD 16
#define HD 64

typedef __attribute__((ext_vector_type(16))) __bf16 v16bf;
typedef __attribute__((ext_vector_type(8)))  __bf16 v8bf;
typedef __attribute__((ext_vector_type(8)))  float  v8f;

__device__ __forceinline__ unsigned short f2bf(float f) {
  unsigned int u = __float_as_uint(f);
  u += 0x7FFFu + ((u >> 16) & 1u);   // round-to-nearest-even
  return (unsigned short)(u >> 16);
}

// Build a 16-bit A-matrix fragment (16x32): lane l holds row l&15,
// K = {koff..koff+15} with koff = (l>=16)?16:0.
// Caller passes p already offset by row*ld + koff; we grab [0..7] and [8..15].
__device__ __forceinline__ v16bf make_afrag(const unsigned short* p) {
  v8bf lo = *(const v8bf*)p;
  v8bf hi = *(const v8bf*)(p + 8);
  return __builtin_shufflevector(lo, hi, 0, 1, 2, 3, 4, 5, 6, 7,
                                 8, 9, 10, 11, 12, 13, 14, 15);
}

__device__ __forceinline__ v8f wmma_bf16(v16bf a, v16bf b, v8f c) {
  return __builtin_amdgcn_wmma_f32_16x16x32_bf16(false, a, false, b,
                                                 (short)0, c, false, false);
}

__global__ void cvt_f32_bf16(const float* __restrict__ in,
                             unsigned short* __restrict__ out, int n) {
  int i = blockIdx.x * blockDim.x + threadIdx.x;
  if (i < n) out[i] = f2bf(in[i]);
}

// Vt[b][h][d][t] = V[(b*SEQ+t)*EMB + h*HD + d]; flat out index == b,h,d,t bitfields
__global__ void transpose_v(const unsigned short* __restrict__ V,
                            unsigned short* __restrict__ Vt) {
  int idx = blockIdx.x * blockDim.x + threadIdx.x;
  int t = idx & (SEQ - 1);
  int d = (idx >> 10) & (HD - 1);
  int h = (idx >> 16) & (NHEAD - 1);
  int b = idx >> 20;
  Vt[idx] = V[((size_t)(b * SEQ + t)) * EMB + h * HD + d];
}

// Y[m][n] = sum_k A[m][k] * W[n][k]   (i.e. Y = A @ W^T), bf16 in, bf16 out.
// Wave tile: 32(M) x 64(N) -> 8 WMMAs per 32-deep k-step against 12 b128 loads.
// Block = 8 waves stacked in M (256 rows). grid = (EMB/64, M/256)
__global__ __launch_bounds__(256) void gemm_bf16_kernel(
    const unsigned short* __restrict__ A, const unsigned short* __restrict__ W,
    unsigned short* __restrict__ Y) {
  const int lane = threadIdx.x & 31;
  const int wave = threadIdx.x >> 5;
  const int hi = lane >> 4, lo = lane & 15;
  const int m0 = (blockIdx.y * 8 + wave) * 32;
  const int n0 = blockIdx.x * 64;

  v8f acc0[4] = {};
  v8f acc1[4] = {};
  const unsigned short* arow0 = A + (size_t)(m0 + lo) * EMB + hi * 16;
  const unsigned short* arow1 = A + (size_t)(m0 + 16 + lo) * EMB + hi * 16;
  const unsigned short* br[4];
#pragma unroll
  for (int j = 0; j < 4; ++j)
    br[j] = W + (size_t)(n0 + j * 16 + lo) * EMB + hi * 16;

  for (int k = 0; k < EMB; k += 32) {
    v16bf a0 = make_afrag(arow0 + k);
    v16bf a1 = make_afrag(arow1 + k);
#pragma unroll
    for (int j = 0; j < 4; ++j) {
      v16bf b = *(const v16bf*)(br[j] + k);
      acc0[j] = wmma_bf16(a0, b, acc0[j]);
      acc1[j] = wmma_bf16(a1, b, acc1[j]);
    }
  }
#pragma unroll
  for (int j = 0; j < 4; ++j)
#pragma unroll
    for (int r = 0; r < 8; ++r) {
      Y[(size_t)(m0 + r + 8 * hi) * EMB + n0 + j * 16 + lo] = f2bf(acc0[j][r]);
      Y[(size_t)(m0 + 16 + r + 8 * hi) * EMB + n0 + j * 16 + lo] =
          f2bf(acc1[j][r]);
    }
}

// Same GEMM, f32 output with bias (final projection into d_out).
__global__ __launch_bounds__(256) void gemm_out_kernel(
    const unsigned short* __restrict__ A, const unsigned short* __restrict__ W,
    const float* __restrict__ bias, float* __restrict__ Y) {
  const int lane = threadIdx.x & 31;
  const int wave = threadIdx.x >> 5;
  const int hi = lane >> 4, lo = lane & 15;
  const int m0 = (blockIdx.y * 8 + wave) * 32;
  const int n0 = blockIdx.x * 64;

  v8f acc0[4] = {};
  v8f acc1[4] = {};
  const unsigned short* arow0 = A + (size_t)(m0 + lo) * EMB + hi * 16;
  const unsigned short* arow1 = A + (size_t)(m0 + 16 + lo) * EMB + hi * 16;
  const unsigned short* br[4];
#pragma unroll
  for (int j = 0; j < 4; ++j)
    br[j] = W + (size_t)(n0 + j * 16 + lo) * EMB + hi * 16;

  for (int k = 0; k < EMB; k += 32) {
    v16bf a0 = make_afrag(arow0 + k);
    v16bf a1 = make_afrag(arow1 + k);
#pragma unroll
    for (int j = 0; j < 4; ++j) {
      v16bf b = *(const v16bf*)(br[j] + k);
      acc0[j] = wmma_bf16(a0, b, acc0[j]);
      acc1[j] = wmma_bf16(a1, b, acc1[j]);
    }
  }
#pragma unroll
  for (int j = 0; j < 4; ++j) {
    float bv = bias[n0 + j * 16 + lo];
#pragma unroll
    for (int r = 0; r < 8; ++r) {
      Y[(size_t)(m0 + r + 8 * hi) * EMB + n0 + j * 16 + lo] = acc0[j][r] + bv;
      Y[(size_t)(m0 + 16 + r + 8 * hi) * EMB + n0 + j * 16 + lo] =
          acc1[j][r] + bv;
    }
  }
}

// Flash attention: one wave = 16 query rows of one (batch, head).
// Streams 32 keys/iter: 4 WMMAs (Q·K^T) + online softmax + 4 WMMAs (P·V).
__global__ __launch_bounds__(256) void attention_kernel(
    const unsigned short* __restrict__ Q, const unsigned short* __restrict__ K,
    const unsigned short* __restrict__ Vt, const int* __restrict__ mask,
    unsigned short* __restrict__ Ctx) {
  __shared__ __align__(32) unsigned short plds[8][16 * 32];
  const int lane = threadIdx.x & 31;
  const int wave = threadIdx.x >> 5;
  const int job = blockIdx.x * 8 + wave;           // NBATCH*NHEAD*(SEQ/16) jobs
  const int qt = (job & 63) * 16;
  const int h = (job >> 6) & 15;
  const int b = job >> 10;
  const int hi = lane >> 4, lo = lane & 15;

  const float SCALE = 0.03125f;                    // 1/sqrt(1024)
  const float L2E = 1.44269504088896f;

  float mrow[8], lrow[8];
  v8f acc[4] = {};
#pragma unroll
  for (int r = 0; r < 8; ++r) { mrow[r] = -3.0e38f; lrow[r] = 0.0f; }

  const unsigned short* qbase =
      Q + ((size_t)(b * SEQ + qt + lo)) * EMB + h * HD + hi * 16;
  v16bf qa0 = make_afrag(qbase);
  v16bf qa1 = make_afrag(qbase + 32);

  const unsigned short* kbase = K + ((size_t)(b * SEQ)) * EMB + h * HD + hi * 16;
  const unsigned short* vtb = Vt + ((size_t)(b * NHEAD + h)) * HD * SEQ;
  const int* mbase = mask + ((size_t)b * SEQ + qt) * SEQ;
  unsigned short* myp = plds[wave];

  for (int kt = 0; kt < SEQ; kt += 32) {
    v8f s0 = {}, s1 = {};
    {
      const unsigned short* k0 = kbase + (size_t)(kt + lo) * EMB;
      const unsigned short* k1 = kbase + (size_t)(kt + 16 + lo) * EMB;
      s0 = wmma_bf16(qa0, *(const v16bf*)k0, s0);
      s0 = wmma_bf16(qa1, *(const v16bf*)(k0 + 32), s0);
      s1 = wmma_bf16(qa0, *(const v16bf*)k1, s1);
      s1 = wmma_bf16(qa1, *(const v16bf*)(k1 + 32), s1);
    }
#pragma unroll
    for (int r = 0; r < 8; ++r) {
      const int q = r + 8 * hi;                    // row index within 16x16 D tile
      const int* mrp = mbase + (size_t)q * SEQ + kt;
      float v0 = (mrp[lo] == 0) ? -1.0e30f : s0[r] * SCALE;
      float v1 = (mrp[16 + lo] == 0) ? -1.0e30f : s1[r] * SCALE;
      float tmax = fmaxf(v0, v1);                  // row max across 16 lanes
      tmax = fmaxf(tmax, __shfl_xor(tmax, 1));
      tmax = fmaxf(tmax, __shfl_xor(tmax, 2));
      tmax = fmaxf(tmax, __shfl_xor(tmax, 4));
      tmax = fmaxf(tmax, __shfl_xor(tmax, 8));
      float mnew = fmaxf(mrow[r], tmax);
      float rescale = exp2f((mrow[r] - mnew) * L2E);
      mrow[r] = mnew;
      float e0 = exp2f((v0 - mnew) * L2E);
      float e1 = exp2f((v1 - mnew) * L2E);
      float rs = e0 + e1;
      rs += __shfl_xor(rs, 1);
      rs += __shfl_xor(rs, 2);
      rs += __shfl_xor(rs, 4);
      rs += __shfl_xor(rs, 8);
      lrow[r] = lrow[r] * rescale + rs;
#pragma unroll
      for (int j = 0; j < 4; ++j) acc[j][r] *= rescale;
      myp[q * 32 + lo] = f2bf(e0);                 // D-layout -> LDS row-major
      myp[q * 32 + 16 + lo] = f2bf(e1);
    }
    // Re-read probabilities in A-fragment layout (same wave; DScnt-ordered).
    v16bf pa = make_afrag(myp + lo * 32 + hi * 16);
#pragma unroll
    for (int j = 0; j < 4; ++j) {
      v16bf bv = *(const v16bf*)(vtb + (size_t)(j * 16 + lo) * SEQ + kt + hi * 16);
      acc[j] = wmma_bf16(pa, bv, acc[j]);
    }
  }

#pragma unroll
  for (int r = 0; r < 8; ++r) {
    float inv = 1.0f / lrow[r];
    const size_t row = (size_t)(b * SEQ + qt + r + 8 * hi);
#pragma unroll
    for (int j = 0; j < 4; ++j)
      Ctx[row * EMB + h * HD + j * 16 + lo] = f2bf(acc[j][r] * inv);
  }
}

extern "C" void kernel_launch(void* const* d_in, const int* in_sizes, int n_in,
                              void* d_out, int out_size, void* d_ws,
                              size_t ws_size, hipStream_t stream) {
  (void)in_sizes; (void)n_in; (void)out_size; (void)ws_size;
  const float* X = (const float*)d_in[0];
  const int* mask = (const int*)d_in[1];
  const float* Wq = (const float*)d_in[2];
  const float* Wk = (const float*)d_in[3];
  const float* Wv = (const float*)d_in[4];
  const float* Wo = (const float*)d_in[5];
  const float* bo = (const float*)d_in[6];
  float* out = (float*)d_out;

  const size_t XB = (size_t)NBATCH * SEQ * EMB;    // 16,777,216 elems
  const size_t WB = (size_t)EMB * EMB;             // 1,048,576 elems

  unsigned short* p = (unsigned short*)d_ws;
  unsigned short* Xb = p;   p += XB;               // dead after QKV; reused as Vt
  unsigned short* Wqb = p;  p += WB;
  unsigned short* Wkb = p;  p += WB;
  unsigned short* Wvb = p;  p += WB;
  unsigned short* Wob = p;  p += WB;
  unsigned short* Qb = p;   p += XB;
  unsigned short* Kb = p;   p += XB;
  unsigned short* Vb = p;   p += XB;               // dead after transpose; reused as Ctx
  unsigned short* Vt = Xb;
  unsigned short* Ctx = Vb;

  cvt_f32_bf16<<<(int)(XB / 256), 256, 0, stream>>>(X, Xb, (int)XB);
  cvt_f32_bf16<<<(int)(WB / 256), 256, 0, stream>>>(Wq, Wqb, (int)WB);
  cvt_f32_bf16<<<(int)(WB / 256), 256, 0, stream>>>(Wk, Wkb, (int)WB);
  cvt_f32_bf16<<<(int)(WB / 256), 256, 0, stream>>>(Wv, Wvb, (int)WB);
  cvt_f32_bf16<<<(int)(WB / 256), 256, 0, stream>>>(Wo, Wob, (int)WB);

  dim3 ggrid(EMB / 64, (NBATCH * SEQ) / 256);
  gemm_bf16_kernel<<<ggrid, 256, 0, stream>>>(Xb, Wqb, Qb);
  gemm_bf16_kernel<<<ggrid, 256, 0, stream>>>(Xb, Wkb, Kb);
  gemm_bf16_kernel<<<ggrid, 256, 0, stream>>>(Xb, Wvb, Vb);

  transpose_v<<<(int)(XB / 256), 256, 0, stream>>>(Vb, Vt);

  attention_kernel<<<(NBATCH * NHEAD * (SEQ / 16)) / 8, 256, 0, stream>>>(
      Qb, Kb, Vt, mask, Ctx);

  gemm_out_kernel<<<ggrid, 256, 0, stream>>>(Ctx, Wob, bo, out);
}
